// GAT_82652350644679
// MI455X (gfx1250) — matrix operand order repaired and checked by reading, not compiled, not run on hardware
//
#include <hip/hip_runtime.h>
#include <math.h>

typedef __attribute__((ext_vector_type(16))) _Float16 v16h;
typedef __attribute__((ext_vector_type(8)))  float    v8f;

#define NN   100000
#define EE   1600000
#define FIN  64
#define HH   128
#define GG   4000
#define EPE  (EE + NN)   // edges incl. self loops

// ---------- helpers ----------
__device__ __forceinline__ float lrelu(float x, float s) { return x > 0.f ? x : s * x; }

// order-preserving float <-> uint key for atomicMax-based segment max
__device__ __forceinline__ unsigned fkey(float f) {
    unsigned u = __float_as_uint(f);
    return (u & 0x80000000u) ? ~u : (u | 0x80000000u);
}
__device__ __forceinline__ float funkey(unsigned k) {
    unsigned u = (k & 0x80000000u) ? (k & 0x7FFFFFFFu) : ~k;
    return __uint_as_float(u);
}

__device__ __forceinline__ float wave_sum32(float v) {
#pragma unroll
    for (int off = 16; off > 0; off >>= 1) v += __shfl_xor(v, off);
    return v;
}

// ---------- WMMA GEMM: C[N x 128] = X[N x K] * W[K x 128], f16 inputs, f32 accum ----------
template <int K>
__global__ __launch_bounds__(256) void gemm_wmma_kernel(const float* __restrict__ X,
                                                        const float* __restrict__ Wm,
                                                        float* __restrict__ Cout) {
    const int wave = threadIdx.x >> 5;        // 8 waves: one 16x16 tile each (cols)
    const int lane = threadIdx.x & 31;
    const int hv   = lane >> 4;               // wave half (0/1)
    const int l16  = lane & 15;
    const int rowBase = blockIdx.x * 16;      // 16 node rows per block
    const int colBase = wave * 16;            // 8 * 16 = 128 cols

    v8f c = {};
#pragma unroll
    for (int kc = 0; kc < K; kc += 32) {
        v16h a, b;
#pragma unroll
        for (int v = 0; v < 8; ++v) {
            // A (16x32, 16-bit) layout: v<4 -> k = 8*half + 2v ; v>=4 -> k = 16 + 8*half + 2(v-4)
            const int k0a = (v < 4) ? (hv * 8 + 2 * v) : (16 + hv * 8 + 2 * (v - 4));
            const int ka  = kc + k0a;
            const float* xr = X + (long)(rowBase + l16) * K + ka;
            a[2 * v]     = (_Float16)xr[0];
            a[2 * v + 1] = (_Float16)xr[1];
            // B (32x16, 16-bit) layout: same as A: v<4 -> k = 8*half + 2v ; v>=4 -> k = 16 + 8*half + 2(v-4)
            const int k0b = (v < 4) ? (hv * 8 + 2 * v) : (16 + hv * 8 + 2 * (v - 4));
            const int kb  = kc + k0b;
            b[2 * v]     = (_Float16)Wm[(long)kb * HH + colBase + l16];
            b[2 * v + 1] = (_Float16)Wm[(long)(kb + 1) * HH + colBase + l16];
        }
        c = __builtin_amdgcn_wmma_f32_16x16x32_f16(false, a, false, b, (short)0, c, false, false);
    }
#pragma unroll
    for (int r = 0; r < 8; ++r) {
        const int m = r + hv * 8;             // C layout: lanes 0-15 M=r, lanes 16-31 M=r+8
        Cout[(long)(rowBase + m) * HH + colBase + l16] = c[r];
    }
}

// ---------- per-node attention logits: s_src[n] = h[n].a_src, s_dst[n] = h[n].a_dst ----------
__global__ __launch_bounds__(256) void node_dots_kernel(const float* __restrict__ hg,
                                                        const float* __restrict__ a_s,
                                                        const float* __restrict__ a_d,
                                                        float* __restrict__ s_src,
                                                        float* __restrict__ s_dst) {
    const int n    = blockIdx.x * 8 + (threadIdx.x >> 5);
    const int lane = threadIdx.x & 31;
    if (n >= NN) return;
    const float4 h4 = ((const float4*)(hg + (long)n * HH))[lane];
    const float4 as = ((const float4*)a_s)[lane];
    const float4 ad = ((const float4*)a_d)[lane];
    float vs = h4.x * as.x + h4.y * as.y + h4.z * as.z + h4.w * as.w;
    float vd = h4.x * ad.x + h4.y * ad.y + h4.z * ad.z + h4.w * ad.w;
    vs = wave_sum32(vs);
    vd = wave_sum32(vd);
    if (lane == 0) { s_src[n] = vs; s_dst[n] = vd; }
}

// ---------- zero agg, init segment max/sum state ----------
__global__ __launch_bounds__(256) void init_layer_kernel(float* __restrict__ agg,
                                                         unsigned* __restrict__ mkey,
                                                         float* __restrict__ ssum) {
    const long i = (long)blockIdx.x * 256 + threadIdx.x;
    if (i < (long)NN * HH) agg[i] = 0.f;
    if (i < NN) { mkey[i] = fkey(-__builtin_inff()); ssum[i] = 0.f; }
}

__device__ __forceinline__ void edge_sd(const int* __restrict__ ei, int e, int& s, int& d) {
    if (e < EE) { s = ei[e]; d = ei[EE + e]; } else { s = e - EE; d = s; }
}

// ---------- pass 1: segment max of leaky_relu logits ----------
__global__ __launch_bounds__(256) void edge_max_kernel(const int* __restrict__ ei,
                                                       const float* __restrict__ s_src,
                                                       const float* __restrict__ s_dst,
                                                       unsigned* __restrict__ mkey) {
    const int e = blockIdx.x * 256 + threadIdx.x;
    if (e >= EPE) return;
    int s, d; edge_sd(ei, e, s, d);
    const float ev = lrelu(s_src[s] + s_dst[d], 0.2f);
    atomicMax(&mkey[d], fkey(ev));
}

// ---------- pass 2: t = exp(e - m[dst]); alpha_raw = t; ssum[dst] += t ----------
__global__ __launch_bounds__(256) void edge_exp_kernel(const int* __restrict__ ei,
                                                       const float* __restrict__ s_src,
                                                       const float* __restrict__ s_dst,
                                                       const unsigned* __restrict__ mkey,
                                                       float* __restrict__ ssum,
                                                       float* __restrict__ alpha) {
    const int e = blockIdx.x * 256 + threadIdx.x;
    if (e >= EPE) return;
    int s, d; edge_sd(ei, e, s, d);
    const float ev = lrelu(s_src[s] + s_dst[d], 0.2f);
    const float mv = funkey(mkey[d]);
    const float t  = __expf(ev - mv);
    alpha[e] = t;
    atomicAdd(&ssum[d], t);
}

// ---------- pass 3: normalize alpha; agg[dst] += alpha * h[src]  (wave per edge) ----------
__global__ __launch_bounds__(256) void edge_agg_kernel(const int* __restrict__ ei,
                                                       const float* __restrict__ hg,
                                                       float* __restrict__ alpha,
                                                       const float* __restrict__ ssum,
                                                       float* __restrict__ agg) {
    const int e    = blockIdx.x * 8 + (threadIdx.x >> 5);
    const int lane = threadIdx.x & 31;
    if (e >= EPE) return;
    int s, d; edge_sd(ei, e, s, d);
    const float t  = alpha[e];                 // whole-wave load before lane0 store
    const float al = t / (ssum[d] + 1e-16f);
    if (lane == 0) alpha[e] = al;
    const float4 h4 = ((const float4*)(hg + (long)s * HH))[lane];
    float* dp = agg + (long)d * HH + lane * 4;
    atomicAdd(dp + 0, h4.x * al);
    atomicAdd(dp + 1, h4.y * al);
    atomicAdd(dp + 2, h4.z * al);
    atomicAdd(dp + 3, h4.w * al);
}

// ---------- bias + layernorm + leaky_relu(0.1)  (wave per node, in-place safe) ----------
__global__ __launch_bounds__(256) void node_ln_kernel(const float* __restrict__ agg,
                                                      const float* __restrict__ bias,
                                                      const float* __restrict__ g,
                                                      const float* __restrict__ be,
                                                      float* __restrict__ outb) {
    const int n    = blockIdx.x * 8 + (threadIdx.x >> 5);
    const int lane = threadIdx.x & 31;
    if (n >= NN) return;
    float4 v  = ((const float4*)(agg + (long)n * HH))[lane];
    const float4 bb = ((const float4*)bias)[lane];
    v.x += bb.x; v.y += bb.y; v.z += bb.z; v.w += bb.w;
    float sum = v.x + v.y + v.z + v.w;
    float sq  = v.x * v.x + v.y * v.y + v.z * v.z + v.w * v.w;
    sum = wave_sum32(sum);
    sq  = wave_sum32(sq);
    const float mu  = sum * (1.f / HH);
    const float var = sq * (1.f / HH) - mu * mu;
    const float rs  = rsqrtf(var + 1e-5f);
    const float4 gg = ((const float4*)g)[lane];
    const float4 bt = ((const float4*)be)[lane];
    float4 o;
    o.x = lrelu((v.x - mu) * rs * gg.x + bt.x, 0.1f);
    o.y = lrelu((v.y - mu) * rs * gg.y + bt.y, 0.1f);
    o.z = lrelu((v.z - mu) * rs * gg.z + bt.z, 0.1f);
    o.w = lrelu((v.w - mu) * rs * gg.w + bt.w, 0.1f);
    ((float4*)(outb + (long)n * HH))[lane] = o;
}

// ---------- pooling ----------
__global__ __launch_bounds__(256) void pool_init_kernel(unsigned* __restrict__ pkey) {
    const int i = blockIdx.x * 256 + threadIdx.x;
    if (i < GG * HH) pkey[i] = fkey(-__builtin_inff());
}

__global__ __launch_bounds__(256) void pool_max_kernel(const float* __restrict__ nodes,
                                                       const int* __restrict__ batch,
                                                       unsigned* __restrict__ pkey) {
    const int n    = blockIdx.x * 8 + (threadIdx.x >> 5);
    const int lane = threadIdx.x & 31;
    if (n >= NN) return;
    const int grp = batch[n];
    const float4 v = ((const float4*)(nodes + (long)n * HH))[lane];
    unsigned* p = pkey + (long)grp * HH + lane * 4;
    atomicMax(p + 0, fkey(v.x));
    atomicMax(p + 1, fkey(v.y));
    atomicMax(p + 2, fkey(v.z));
    atomicMax(p + 3, fkey(v.w));
}

__global__ __launch_bounds__(256) void head_kernel(const unsigned* __restrict__ pkey,
                                                   const float* __restrict__ fcW,
                                                   const float* __restrict__ fcb,
                                                   float* __restrict__ out) {
    const int grp  = blockIdx.x * 8 + (threadIdx.x >> 5);
    const int lane = threadIdx.x & 31;
    if (grp >= GG) return;
    const uint4 k = ((const uint4*)(pkey + (long)grp * HH))[lane];
    float4 v;
    v.x = fmaxf(funkey(k.x), 0.f);
    v.y = fmaxf(funkey(k.y), 0.f);
    v.z = fmaxf(funkey(k.z), 0.f);
    v.w = fmaxf(funkey(k.w), 0.f);
    const float4 w = ((const float4*)fcW)[lane];
    float dot = v.x * w.x + v.y * w.y + v.z * w.z + v.w * w.w;
    dot = wave_sum32(dot);
    if (lane == 0) out[grp] = dot + fcb[0];
}

// ---------- launch ----------
extern "C" void kernel_launch(void* const* d_in, const int* in_sizes, int n_in,
                              void* d_out, int out_size, void* d_ws, size_t ws_size,
                              hipStream_t stream) {
    const float* x      = (const float*)d_in[0];
    const int*   ei     = (const int*)d_in[1];
    const int*   batch  = (const int*)d_in[3];
    const float* W1     = (const float*)d_in[4];
    const float* a_src1 = (const float*)d_in[5];
    const float* a_dst1 = (const float*)d_in[6];
    const float* b1     = (const float*)d_in[7];
    const float* W2     = (const float*)d_in[8];
    const float* a_src2 = (const float*)d_in[9];
    const float* a_dst2 = (const float*)d_in[10];
    const float* b2     = (const float*)d_in[11];
    const float* ln_g   = (const float*)d_in[12];
    const float* ln_b   = (const float*)d_in[13];
    const float* fcW    = (const float*)d_in[14];
    const float* fcb    = (const float*)d_in[15];

    float* out    = (float*)d_out;     // [G]
    float* alpha1 = out + GG;          // [E']
    float* alpha2 = alpha1 + EPE;      // [E']

    float* ws     = (float*)d_ws;
    float* A      = ws;                         // N*H : GEMM output (both layers)
    float* Bf     = A + (size_t)NN * HH;        // N*H : agg / LN out / layer-2 input
    float* s_src  = Bf + (size_t)NN * HH;
    float* s_dst  = s_src + NN;
    float* ssum   = s_dst + NN;
    unsigned* mkey = (unsigned*)(ssum + NN);
    unsigned* pkey = mkey + NN;                 // G*H

    const dim3 blk(256);
    const int gGemm  = NN / 16;                 // 6250
    const int gNode  = NN / 8;                  // 12500 (wave per node)
    const int gInit  = (NN * HH) / 256;         // 50000
    const int gEdge  = (EPE + 255) / 256;
    const int gEdgeW = (EPE + 7) / 8;           // wave per edge

    // ---- layer 1 ----
    gemm_wmma_kernel<FIN><<<gGemm, blk, 0, stream>>>(x, W1, A);
    node_dots_kernel<<<gNode, blk, 0, stream>>>(A, a_src1, a_dst1, s_src, s_dst);
    init_layer_kernel<<<gInit, blk, 0, stream>>>(Bf, mkey, ssum);
    edge_max_kernel<<<gEdge, blk, 0, stream>>>(ei, s_src, s_dst, mkey);
    edge_exp_kernel<<<gEdge, blk, 0, stream>>>(ei, s_src, s_dst, mkey, ssum, alpha1);
    edge_agg_kernel<<<gEdgeW, blk, 0, stream>>>(ei, A, alpha1, ssum, Bf);
    node_ln_kernel<<<gNode, blk, 0, stream>>>(Bf, b1, ln_g, ln_b, Bf);

    // ---- layer 2 ----
    gemm_wmma_kernel<HH><<<gGemm, blk, 0, stream>>>(Bf, W2, A);
    node_dots_kernel<<<gNode, blk, 0, stream>>>(A, a_src2, a_dst2, s_src, s_dst);
    init_layer_kernel<<<gInit, blk, 0, stream>>>(Bf, mkey, ssum);   // Bf free after GEMM2
    edge_max_kernel<<<gEdge, blk, 0, stream>>>(ei, s_src, s_dst, mkey);
    edge_exp_kernel<<<gEdge, blk, 0, stream>>>(ei, s_src, s_dst, mkey, ssum, alpha2);
    edge_agg_kernel<<<gEdgeW, blk, 0, stream>>>(ei, A, alpha2, ssum, Bf);
    node_ln_kernel<<<gNode, blk, 0, stream>>>(Bf, b2, ln_g, ln_b, Bf);

    // ---- pool + head ----
    pool_init_kernel<<<(GG * HH + 255) / 256, blk, 0, stream>>>(pkey);
    pool_max_kernel<<<gNode, blk, 0, stream>>>(Bf, batch, pkey);
    head_kernel<<<(GG + 7) / 8, blk, 0, stream>>>(pkey, fcW, fcb, out);
}
